// MultimodalDeformableTransformer_85074712199412
// MI455X (gfx1250) — compile-verified, matrix operand order checked by reading, not run on hardware
//
#include <hip/hip_runtime.h>
#include <hip/hip_bf16.h>
#include <cstdint>

// ---------------------------------------------------------------------------
// Multimodal deformable transformer decoder (6 layers) for gfx1250 / MI455X.
// GEMMs: v_wmma_f32_16x16x32_f16, 128-wide tiles, double-buffered LDS.
// Big value-projection GEMMs use the Tensor Data Mover (tensor_load_to_lds)
// when the builtin is available; s_wait_tensorcnt gates the pipeline.
// ---------------------------------------------------------------------------

typedef _Float16 half_t;
typedef __attribute__((ext_vector_type(16))) _Float16 v16h;
typedef __attribute__((ext_vector_type(8)))  float    v8f;
typedef __attribute__((ext_vector_type(4)))  unsigned int v4u;
typedef __attribute__((ext_vector_type(8)))  int      v8i;
typedef __attribute__((ext_vector_type(4)))  int      v4i;

#define BATCH  32
#define NQ     300
#define MROWS  (BATCH * NQ)          // 9600
#define LV     3840
#define LA     1920
#define NLAYER 6

#if defined(__HIP_DEVICE_COMPILE__) && __has_builtin(__builtin_amdgcn_tensor_load_to_lds)
#define TDM_AVAILABLE 1
#else
#define TDM_AVAILABLE 0
#endif

// ---------------------------------------------------------------------------
// WMMA helpers
// ---------------------------------------------------------------------------
__device__ __forceinline__ v8f wmma_f16(v16h a, v16h b, v8f c) {
  return __builtin_amdgcn_wmma_f32_16x16x32_f16(false, a, false, b, (short)0,
                                                c, false, false);
}

// ISA 7.12.2 16-bit 16x32 fragment: lane (hi,l16); VGPR j holds
// K = (j>=4?16:0) + hi*8 + 2*(j&3) (+1). Collapses to 2x ds_load_b128.
__device__ __forceinline__ v16h frag_f16(const half_t* row, int hi) {
  v16h f;
#pragma unroll
  for (int j = 0; j < 8; ++j) {
    const int kb = ((j & 4) << 2) + hi * 8 + ((j & 3) << 1);
    f[2 * j]     = row[kb];
    f[2 * j + 1] = row[kb + 1];
  }
  return f;
}

__device__ __forceinline__ v16h frag_f32(const float* row, int hi) {
  v16h f;
#pragma unroll
  for (int j = 0; j < 8; ++j) {
    const int kb = ((j & 4) << 2) + hi * 8 + ((j & 3) << 1);
    f[2 * j]     = (half_t)row[kb];
    f[2 * j + 1] = (half_t)row[kb + 1];
  }
  return f;
}

// ---------------------------------------------------------------------------
// TDM: load a rows x 32-float tile (row stride K floats) into LDS.
// D# per ISA 8.3-8.5: group0 = flags/lds_addr/global_addr/type=2,
// group1 = data_size=4B, tensor/tile dims, dim0 stride.
// ---------------------------------------------------------------------------
#if TDM_AVAILABLE
__device__ __forceinline__ void tdm_load_tile(const float* gsrc,
                                              unsigned lds_off, int rows,
                                              int K) {
  const unsigned long long ga = (unsigned long long)(uintptr_t)gsrc;
  v4u g0;
  g0[0] = 1u;                                    // count=1, flags 0
  g0[1] = lds_off;                               // lds_addr [63:32]
  g0[2] = (unsigned)ga;                          // global_addr [95:64]
  g0[3] = (unsigned)((ga >> 32) & 0x01ffffffu) | (2u << 30);  // addr hi + type=2
  v8i g1;
  g1[0] = (2 << 16);                 // data_size = 4 bytes
  g1[1] = (32 << 16);                // tensor_dim0 = 32  (bits 63:48)
  g1[2] = (rows & 0xffff) << 16;     // tensor_dim1 = rows (bits 95:80)
  g1[3] = (32 << 16);                // tile_dim0 = 32 (bits 127:112)
  g1[4] = rows;                      // tile_dim1 = rows (bits 143:128)
  g1[5] = K;                         // tensor_dim0_stride lo
  g1[6] = 0;
  g1[7] = 0;
  v4i gz = {0, 0, 0, 0};
#if __has_include(<hip/amd_detail/amd_gfx1250_TDM.h>)
  v8i gz8 = {0, 0, 0, 0, 0, 0, 0, 0};
  __builtin_amdgcn_tensor_load_to_lds(g0, g1, gz, gz, gz8, 0);
#else
  __builtin_amdgcn_tensor_load_to_lds(g0, g1, gz, gz, 0);
#endif
}

__device__ __forceinline__ void tdm_wait0() {
#if __has_builtin(__builtin_amdgcn_s_wait_tensorcnt)
  __builtin_amdgcn_s_wait_tensorcnt(0);
#else
  asm volatile("s_wait_tensorcnt 0x0" ::: "memory");
#endif
}
#endif  // TDM_AVAILABLE

// ---------------------------------------------------------------------------
// GEMM:  C[M,N] = A[M,K] @ W[N,K]^T + bias[N]  (optional ReLU)
// Block tile 128x128, BK=32, f16 double-buffered LDS, 1 barrier / K chunk.
// 8 waves: wave = (wm 0..3: 32 rows) x (wn 0..1: 64 cols) -> 8 WMMAs/chunk.
// ---------------------------------------------------------------------------
template <int RELU>
__global__ __launch_bounds__(256) void gemm_wmma_kernel(
    const float* __restrict__ A, const float* __restrict__ W,
    const float* __restrict__ bias, float* __restrict__ C,
    int M, int N, int K) {
  __shared__ half_t As[2][128][32];
  __shared__ half_t Ws[2][128][32];

  const int tid   = threadIdx.x;
  const int mBase = blockIdx.x * 128;
  const int nBase = blockIdx.y * 128;
  const int lane  = tid & 31;
  const int wave  = tid >> 5;
  const int wm    = wave & 3;
  const int wn    = wave >> 2;
  const int hi    = lane >> 4;
  const int l16   = lane & 15;

  v8f acc[2][4] = {};

  const int sr = tid >> 1;          // staging row 0..127
  const int sc = (tid & 1) * 16;    // staging col 0 / 16

  const float* aptr = A + (size_t)(mBase + sr) * K + sc;
  const float* wptr = W + (size_t)(nBase + sr) * K + sc;

  float ar[16], wr[16];
#pragma unroll
  for (int i = 0; i < 4; ++i) {
    float4 av = *(const float4*)(aptr + 4 * i);
    float4 wv = *(const float4*)(wptr + 4 * i);
    ar[4 * i] = av.x; ar[4 * i + 1] = av.y; ar[4 * i + 2] = av.z; ar[4 * i + 3] = av.w;
    wr[4 * i] = wv.x; wr[4 * i + 1] = wv.y; wr[4 * i + 2] = wv.z; wr[4 * i + 3] = wv.w;
  }
#pragma unroll
  for (int i = 0; i < 16; ++i) {
    As[0][sr][sc + i] = (half_t)ar[i];
    Ws[0][sr][sc + i] = (half_t)wr[i];
  }

  const int nIter = K / 32;
  int buf = 0;
  for (int it = 0; it < nIter; ++it) {
    __syncthreads();
    const bool more = (it + 1) < nIter;
    if (more) {  // global loads for next chunk overlap the WMMAs below
      const float* ap = aptr + (it + 1) * 32;
      const float* wp = wptr + (it + 1) * 32;
      __builtin_prefetch(ap + 32, 0, 3);
      __builtin_prefetch(wp + 32, 0, 3);
#pragma unroll
      for (int i = 0; i < 4; ++i) {
        float4 av = *(const float4*)(ap + 4 * i);
        float4 wv = *(const float4*)(wp + 4 * i);
        ar[4 * i] = av.x; ar[4 * i + 1] = av.y; ar[4 * i + 2] = av.z; ar[4 * i + 3] = av.w;
        wr[4 * i] = wv.x; wr[4 * i + 1] = wv.y; wr[4 * i + 2] = wv.z; wr[4 * i + 3] = wv.w;
      }
    }
    const v16h a0 = frag_f16(&As[buf][wm * 32 + l16][0], hi);
    const v16h a1 = frag_f16(&As[buf][wm * 32 + 16 + l16][0], hi);
#pragma unroll
    for (int j = 0; j < 4; ++j) {
      const v16h bf = frag_f16(&Ws[buf][wn * 64 + j * 16 + l16][0], hi);
      acc[0][j] = wmma_f16(a0, bf, acc[0][j]);
      acc[1][j] = wmma_f16(a1, bf, acc[1][j]);
    }
    if (more) {
#pragma unroll
      for (int i = 0; i < 16; ++i) {
        As[buf ^ 1][sr][sc + i] = (half_t)ar[i];
        Ws[buf ^ 1][sr][sc + i] = (half_t)wr[i];
      }
    }
    buf ^= 1;
  }

  const int col00 = nBase + wn * 64 + l16;
#pragma unroll
  for (int i = 0; i < 2; ++i) {
    const int row0 = mBase + wm * 32 + i * 16 + hi * 8;
#pragma unroll
    for (int j = 0; j < 4; ++j) {
      const int col = col00 + j * 16;
      const float bb = bias[col];
#pragma unroll
      for (int r = 0; r < 8; ++r) {
        float v = acc[i][j][r] + bb;
        if (RELU) v = fmaxf(v, 0.f);
        C[(size_t)(row0 + r) * N + col] = v;
      }
    }
  }
}

// ---------------------------------------------------------------------------
// TDM-staged GEMM (used for the big value-projection GEMMs).
// Block tile 128x64, f32 double-buffered LDS filled by the Tensor Data Mover;
// wave 0 is the producer (issue after barrier, s_wait_tensorcnt before it).
// 8 waves: (wm 0..3: 32 rows) x (wn 0..1: 32 cols) -> 4 WMMAs/chunk.
// ---------------------------------------------------------------------------
__global__ __launch_bounds__(256) void gemm_wmma_tdm_kernel(
    const float* __restrict__ A, const float* __restrict__ W,
    const float* __restrict__ bias, float* __restrict__ C,
    int M, int N, int K) {
  __shared__ float Asf[2][128][32];
  __shared__ float Wsf[2][64][32];

  const int tid   = threadIdx.x;
  const int mBase = blockIdx.x * 128;
  const int nBase = blockIdx.y * 64;
  const int lane  = tid & 31;
  const int wave  = tid >> 5;
  const int wm    = wave & 3;
  const int wn    = wave >> 2;
  const int hi    = lane >> 4;
  const int l16   = lane & 15;

  v8f acc[2][2] = {};
  const int nIter = K / 32;

#if TDM_AVAILABLE
  const bool issuer = (wave == 0);
  if (issuer) {
    tdm_load_tile(A + (size_t)mBase * K, (unsigned)(size_t)&Asf[0][0][0], 128, K);
    tdm_load_tile(W + (size_t)nBase * K, (unsigned)(size_t)&Wsf[0][0][0], 64, K);
  }
  int buf = 0;
  for (int it = 0; it < nIter; ++it) {
    if (issuer) tdm_wait0();
    __syncthreads();
    if (issuer && (it + 1) < nIter) {  // DMA next chunk while WMMAs run
      const int k0 = (it + 1) * 32;
      tdm_load_tile(A + (size_t)mBase * K + k0,
                    (unsigned)(size_t)&Asf[buf ^ 1][0][0], 128, K);
      tdm_load_tile(W + (size_t)nBase * K + k0,
                    (unsigned)(size_t)&Wsf[buf ^ 1][0][0], 64, K);
    }
    const v16h a0 = frag_f32(&Asf[buf][wm * 32 + l16][0], hi);
    const v16h a1 = frag_f32(&Asf[buf][wm * 32 + 16 + l16][0], hi);
#pragma unroll
    for (int j = 0; j < 2; ++j) {
      const v16h bf = frag_f32(&Wsf[buf][wn * 32 + j * 16 + l16][0], hi);
      acc[0][j] = wmma_f16(a0, bf, acc[0][j]);
      acc[1][j] = wmma_f16(a1, bf, acc[1][j]);
    }
    buf ^= 1;
  }
#else
  // Fallback staging without the TDM builtin (cooperative loads, 1 buffer)
  for (int it = 0; it < nIter; ++it) {
    const int k0 = it * 32;
    for (int idx = tid; idx < 128 * 32; idx += 256) {
      const int r = idx >> 5, c = idx & 31;
      Asf[0][r][c] = A[(size_t)(mBase + r) * K + k0 + c];
    }
    for (int idx = tid; idx < 64 * 32; idx += 256) {
      const int r = idx >> 5, c = idx & 31;
      Wsf[0][r][c] = W[(size_t)(nBase + r) * K + k0 + c];
    }
    __syncthreads();
    const v16h a0 = frag_f32(&Asf[0][wm * 32 + l16][0], hi);
    const v16h a1 = frag_f32(&Asf[0][wm * 32 + 16 + l16][0], hi);
#pragma unroll
    for (int j = 0; j < 2; ++j) {
      const v16h bf = frag_f32(&Wsf[0][wn * 32 + j * 16 + l16][0], hi);
      acc[0][j] = wmma_f16(a0, bf, acc[0][j]);
      acc[1][j] = wmma_f16(a1, bf, acc[1][j]);
    }
    __syncthreads();
  }
#endif

  const int col00 = nBase + wn * 32 + l16;
#pragma unroll
  for (int i = 0; i < 2; ++i) {
    const int row0 = mBase + wm * 32 + i * 16 + hi * 8;
#pragma unroll
    for (int j = 0; j < 2; ++j) {
      const int col = col00 + j * 16;
      const float bb = bias[col];
#pragma unroll
      for (int r = 0; r < 8; ++r) {
        C[(size_t)(row0 + r) * N + col] = acc[i][j][r] + bb;
      }
    }
  }
}

// ---------------------------------------------------------------------------
// LayerNorm over WIDTH (256/512): out = LN(x [+ res]) * g + b
// ---------------------------------------------------------------------------
template <int WIDTH>
__global__ __launch_bounds__(256) void ln_kernel(
    float* __restrict__ out, int ostride,
    const float* __restrict__ x, const float* __restrict__ res,
    const float* __restrict__ g, const float* __restrict__ b) {
  constexpr int EPT = WIDTH / 256;
  __shared__ float s1[256];
  __shared__ float s2[256];
  const int row = blockIdx.x;
  const int tid = threadIdx.x;

  float v[EPT];
  float sum = 0.f, sq = 0.f;
#pragma unroll
  for (int e = 0; e < EPT; ++e) {
    const int c = tid + e * 256;
    float t = x[(size_t)row * WIDTH + c];
    if (res) t += res[(size_t)row * WIDTH + c];
    v[e] = t;
    sum += t;
    sq += t * t;
  }
  s1[tid] = sum;
  s2[tid] = sq;
  __syncthreads();
  for (int s = 128; s > 0; s >>= 1) {
    if (tid < s) { s1[tid] += s1[tid + s]; s2[tid] += s2[tid + s]; }
    __syncthreads();
  }
  const float mean = s1[0] * (1.0f / WIDTH);
  const float var  = s2[0] * (1.0f / WIDTH) - mean * mean;
  const float inv  = rsqrtf(var + 1e-5f);
#pragma unroll
  for (int e = 0; e < EPT; ++e) {
    const int c = tid + e * 256;
    out[(size_t)row * ostride + c] = (v[e] - mean) * inv * g[c] + b[c];
  }
}

__global__ void addqpos_kernel(float* __restrict__ out,
                               const float* __restrict__ tgt,
                               const float* __restrict__ qpos, int total) {
  const int i = blockIdx.x * blockDim.x + threadIdx.x;
  if (i >= total) return;
  const int row = i >> 8;
  const int c   = i & 255;
  out[i] = tgt[i] + qpos[((row % NQ) << 8) + c];
}

__global__ void copy_kernel(float* __restrict__ dst,
                            const float* __restrict__ src, int n) {
  const int i = blockIdx.x * blockDim.x + threadIdx.x;
  if (i < n) dst[i] = src[i];
}

// ---------------------------------------------------------------------------
// MHA: block per (b,h); thread-per-query online softmax over 300 keys.
// ---------------------------------------------------------------------------
__global__ __launch_bounds__(320) void mha_kernel(
    const float* __restrict__ qk, const float* __restrict__ vproj,
    float* __restrict__ out) {
  const int bh  = blockIdx.x;
  const int b   = bh >> 3;
  const int h   = bh & 7;
  const int tid = threadIdx.x;
  __shared__ float Ks[64][33];
  __shared__ float Vs[64][33];

  float Qf[32], acc[32];
  float mx = -1e30f, ssum = 0.f;
  const bool active = tid < NQ;
  if (active) {
    const size_t qbase = ((size_t)(b * NQ + tid)) * 512 + h * 32;
#pragma unroll
    for (int d = 0; d < 32; ++d) {
      Qf[d]  = qk[qbase + d] * 0.17677669529663687f;  // 1/sqrt(32)
      acc[d] = 0.f;
    }
  }

  for (int k0 = 0; k0 < NQ; k0 += 64) {
    const int cnt = (NQ - k0 < 64) ? (NQ - k0) : 64;
    for (int idx = tid; idx < cnt * 32; idx += 320) {
      const int r = idx >> 5, c = idx & 31;
      const size_t kr = (size_t)(b * NQ + k0 + r);
      Ks[r][c] = qk[kr * 512 + 256 + h * 32 + c];
      Vs[r][c] = vproj[kr * 256 + h * 32 + c];
    }
    __syncthreads();
    if (active) {
      for (int r = 0; r < cnt; ++r) {
        float s = 0.f;
#pragma unroll
        for (int d = 0; d < 32; ++d) s += Qf[d] * Ks[r][d];
        const float mn   = fmaxf(mx, s);
        const float corr = __expf(mx - mn);
        const float e    = __expf(s - mn);
        ssum = ssum * corr + e;
#pragma unroll
        for (int d = 0; d < 32; ++d) acc[d] = acc[d] * corr + e * Vs[r][d];
        mx = mn;
      }
    }
    __syncthreads();
  }
  if (active) {
    const float inv = 1.f / ssum;
    const size_t ob = ((size_t)(b * NQ + tid)) * 256 + h * 32;
#pragma unroll
    for (int d = 0; d < 32; ++d) out[ob + d] = acc[d] * inv;
  }
}

__global__ void softmax16_kernel(float* __restrict__ a, int rows) {
  const int r = blockIdx.x * blockDim.x + threadIdx.x;
  if (r >= rows) return;
  float* p = a + (size_t)r * 16;
  float mx = -1e30f;
#pragma unroll
  for (int i = 0; i < 16; ++i) mx = fmaxf(mx, p[i]);
  float s = 0.f;
  float e[16];
#pragma unroll
  for (int i = 0; i < 16; ++i) { e[i] = __expf(p[i] - mx); s += e[i]; }
  const float inv = 1.f / s;
#pragma unroll
  for (int i = 0; i < 16; ++i) p[i] = e[i] * inv;
}

// ---------------------------------------------------------------------------
// 1D multi-scale deformable sampling. Block per (b,q); lane -> (h, d).
// ---------------------------------------------------------------------------
__global__ __launch_bounds__(256) void deform_kernel(
    float* __restrict__ out, const float* __restrict__ off,
    const float* __restrict__ aw, const float* __restrict__ val,
    const float* __restrict__ refp, const float* __restrict__ vr,
    int T0, int T1, int T2, int T3, int S0, int S1, int S2, int S3, int Lin) {
  const int m   = blockIdx.x;
  const int b   = m / NQ;
  const int tid = threadIdx.x;
  const int h   = tid >> 5;
  const int d   = tid & 31;
  const float rp = refp[m];
  const int Ts[4] = {T0, T1, T2, T3};
  const int Ss[4] = {S0, S1, S2, S3};

  const float* offr = off + (size_t)m * 128 + h * 16;
  const float* awr  = aw + (size_t)m * 128 + h * 16;
  float acc = 0.f;
#pragma unroll
  for (int l = 0; l < 4; ++l) {
    const int   T    = Ts[l];
    const float base = rp * vr[b * 4 + l] * (float)T - 0.5f;
    const float* vl  = val + ((size_t)b * Lin + Ss[l]) * 256 + h * 32 + d;
#pragma unroll
    for (int p = 0; p < 4; ++p) {
      const float pos = base + offr[l * 4 + p];
      const float i0f = floorf(pos);
      const int   i0  = (int)i0f;
      const float w1  = pos - i0f;
      const int   i1  = i0 + 1;
      const float g0 = (i0 >= 0 && i0 < T) ? vl[(size_t)i0 * 256] : 0.f;
      const float g1 = (i1 >= 0 && i1 < T) ? vl[(size_t)i1 * 256] : 0.f;
      acc += awr[l * 4 + p] * (g0 * (1.f - w1) + g1 * w1);
    }
  }
  out[(size_t)m * 256 + h * 32 + d] = acc;
}

// ---------------------------------------------------------------------------
// Host driver
// ---------------------------------------------------------------------------
extern "C" void kernel_launch(void* const* d_in, const int* in_sizes, int n_in,
                              void* d_out, int out_size, void* d_ws,
                              size_t ws_size, hipStream_t stream) {
  (void)in_sizes; (void)n_in; (void)out_size; (void)ws_size;

  const float* tgt0   = (const float*)d_in[0];
  const float* refpts = (const float*)d_in[1];
  const float* qpos   = (const float*)d_in[2];
  const float* vsrc   = (const float*)d_in[4];
  const float* vvr    = (const float*)d_in[7];
  const float* asrc   = (const float*)d_in[9];
  const float* avr    = (const float*)d_in[12];
  const float* w_off  = (const float*)d_in[14];
  const float* b_off  = (const float*)d_in[15];
  const float* w_attn = (const float*)d_in[16];
  const float* b_attn = (const float*)d_in[17];
  const float* w_val  = (const float*)d_in[18];
  const float* b_val  = (const float*)d_in[19];
  const float* w_out  = (const float*)d_in[20];
  const float* b_out  = (const float*)d_in[21];
  const float* w_qkv  = (const float*)d_in[22];
  const float* b_qkv  = (const float*)d_in[23];
  const float* w_o    = (const float*)d_in[24];
  const float* b_o    = (const float*)d_in[25];
  const float* ln1_g  = (const float*)d_in[26];
  const float* ln1_b  = (const float*)d_in[27];
  const float* ln2_g  = (const float*)d_in[28];
  const float* ln2_b  = (const float*)d_in[29];
  const float* ln3_g  = (const float*)d_in[30];
  const float* ln3_b  = (const float*)d_in[31];
  const float* ln4_g  = (const float*)d_in[32];
  const float* ln4_b  = (const float*)d_in[33];
  const float* w1     = (const float*)d_in[34];
  const float* b1     = (const float*)d_in[35];
  const float* w2     = (const float*)d_in[36];
  const float* b2     = (const float*)d_in[37];
  const float* w3     = (const float*)d_in[38];
  const float* b3     = (const float*)d_in[39];

  float* base = (float*)d_ws;
  size_t o = 0;
  auto alloc = [&](size_t n) { float* r = base + o; o += n; return r; };
  float* tgtb  = alloc((size_t)MROWS * 256);
  float* qb    = alloc((size_t)MROWS * 256);
  float* qkb   = alloc((size_t)MROWS * 512);
  float* vb    = alloc((size_t)MROWS * 256);
  float* t2raw = alloc((size_t)MROWS * 256);
  float* t2    = alloc((size_t)MROWS * 256);
  float* valv  = alloc((size_t)BATCH * LV * 256);
  float* vala  = alloc((size_t)BATCH * LA * 256);
  float* offb  = alloc((size_t)MROWS * 128);
  float* awb   = alloc((size_t)MROWS * 128);
  float* tvraw = alloc((size_t)MROWS * 256);
  float* projt = alloc((size_t)MROWS * 256);
  float* xcat  = alloc((size_t)MROWS * 512);
  float* x256  = alloc((size_t)MROWS * 256);
  float* ffn1  = alloc((size_t)MROWS * 1024);
  float* ffn2  = alloc((size_t)MROWS * 256);

  auto gemm = [&](const float* A, const float* W, const float* bias, float* C,
                  int M, int N, int K, bool relu) {
    dim3 g(M / 128, (N + 127) / 128);
    if (relu)
      gemm_wmma_kernel<1><<<g, 256, 0, stream>>>(A, W, bias, C, M, N, K);
    else
      gemm_wmma_kernel<0><<<g, 256, 0, stream>>>(A, W, bias, C, M, N, K);
  };
  auto gemm_big = [&](const float* A, const float* W, const float* bias,
                      float* C, int M, int N, int K) {
    dim3 g(M / 128, N / 64);
    gemm_wmma_tdm_kernel<<<g, 256, 0, stream>>>(A, W, bias, C, M, N, K);
  };

  const int total = MROWS * 256;
  copy_kernel<<<(total + 255) / 256, 256, 0, stream>>>(tgtb, tgt0, total);

  for (int lid = 0; lid < NLAYER; ++lid) {
    const float* Wqkv = w_qkv + (size_t)lid * 768 * 256;
    const float* Bqkv = b_qkv + (size_t)lid * 768;
    const float* Wo   = w_o + (size_t)lid * 256 * 256;
    const float* Bo   = b_o + (size_t)lid * 256;
    const float* Wval = w_val + (size_t)lid * 256 * 256;
    const float* Bval = b_val + (size_t)lid * 256;
    const float* Woff = w_off + (size_t)lid * 128 * 256;
    const float* Boff = b_off + (size_t)lid * 128;
    const float* Watt = w_attn + (size_t)lid * 128 * 256;
    const float* Batt = b_attn + (size_t)lid * 128;
    const float* Wout = w_out + (size_t)lid * 256 * 256;
    const float* Bout = b_out + (size_t)lid * 256;

    // ---- self attention ----
    addqpos_kernel<<<(total + 255) / 256, 256, 0, stream>>>(qb, tgtb, qpos, total);
    gemm(qb, Wqkv, Bqkv, qkb, MROWS, 512, 256, false);                             // Q|K
    gemm(tgtb, Wqkv + (size_t)512 * 256, Bqkv + 512, vb, MROWS, 256, 256, false);  // V
    mha_kernel<<<BATCH * 8, 320, 0, stream>>>(qkb, vb, t2raw);
    gemm(t2raw, Wo, Bo, t2, MROWS, 256, 256, false);
    ln_kernel<256><<<MROWS, 256, 0, stream>>>(tgtb, 256, t2, tgtb,
                                              ln2_g + lid * 256, ln2_b + lid * 256);

    addqpos_kernel<<<(total + 255) / 256, 256, 0, stream>>>(qb, tgtb, qpos, total);

    // ---- video deformable branch ----
    gemm_big(vsrc, Wval, Bval, valv, BATCH * LV, 256, 256);
    gemm(qb, Woff, Boff, offb, MROWS, 128, 256, false);
    gemm(qb, Watt, Batt, awb, MROWS, 128, 256, false);
    softmax16_kernel<<<MROWS * 8 / 256, 256, 0, stream>>>(awb, MROWS * 8);
    deform_kernel<<<MROWS, 256, 0, stream>>>(tvraw, offb, awb, valv, refpts, vvr,
                                             2048, 1024, 512, 256,
                                             0, 2048, 3072, 3584, LV);
    gemm(tvraw, Wout, Bout, projt, MROWS, 256, 256, false);
    ln_kernel<256><<<MROWS, 256, 0, stream>>>(xcat, 512, projt, tgtb,
                                              ln1_g + lid * 256, ln1_b + lid * 256);

    // ---- audio deformable branch ----
    gemm_big(asrc, Wval, Bval, vala, BATCH * LA, 256, 256);
    gemm(qb, Woff, Boff, offb, MROWS, 128, 256, false);
    gemm(qb, Watt, Batt, awb, MROWS, 128, 256, false);
    softmax16_kernel<<<MROWS * 8 / 256, 256, 0, stream>>>(awb, MROWS * 8);
    deform_kernel<<<MROWS, 256, 0, stream>>>(tvraw, offb, awb, vala, refpts, avr,
                                             1024, 512, 256, 128,
                                             0, 1024, 1536, 1792, LA);
    gemm(tvraw, Wout, Bout, projt, MROWS, 256, 256, false);
    ln_kernel<256><<<MROWS, 256, 0, stream>>>(xcat + 256, 512, projt, tgtb,
                                              ln1_g + lid * 256, ln1_b + lid * 256);

    // ---- fuse + FFN ----
    ln_kernel<512><<<MROWS, 256, 0, stream>>>(xcat, 512, xcat, nullptr,
                                              ln4_g + lid * 512, ln4_b + lid * 512);
    gemm(xcat, w3 + (size_t)lid * 256 * 512, b3 + lid * 256, x256,
         MROWS, 256, 512, true);
    gemm(x256, w1 + (size_t)lid * 1024 * 256, b1 + lid * 1024, ffn1,
         MROWS, 1024, 256, true);
    gemm(ffn1, w2 + (size_t)lid * 256 * 1024, b2 + lid * 256, ffn2,
         MROWS, 256, 1024, false);
    ln_kernel<256><<<MROWS, 256, 0, stream>>>(tgtb, 256, ffn2, x256,
                                              ln3_g + lid * 256, ln3_b + lid * 256);
  }

  float* outp = (float*)d_out;
  copy_kernel<<<(total + 255) / 256, 256, 0, stream>>>(outp, tgtb, total);
  copy_kernel<<<(MROWS + 255) / 256, 256, 0, stream>>>(outp + total, refpts, MROWS);
}
